// MultiScaleTimeMixer_52553219834526
// MI455X (gfx1250) — matrix operand order repaired and checked by reading, not compiled, not run on hardware
//
#include <hip/hip_runtime.h>

typedef __attribute__((ext_vector_type(16))) __bf16 v16bf;
typedef __attribute__((ext_vector_type(8)))  float  v8f;

// ---------------- workspace layout (bytes) ----------------
// Each region: hi-plane (N elems bf16) immediately followed by lo-plane (N elems).
constexpr size_t OFF_W0A = 0;        // 128x128 : 16384+16384 bf16 = 65536 B
constexpr size_t OFF_W0B = 65536;    // 128x128
constexpr size_t OFF_C1  = 131072;   // 128x256 : 32768+32768 bf16 = 131072 B
constexpr size_t OFF_C2  = 262144;   // 128x512 : 65536+65536 bf16 = 262144 B
constexpr size_t OFF_W1A = 524288;   // 64x64   : 16384 B
constexpr size_t OFF_W1B = 540672;
constexpr size_t OFF_W2A = 557056;   // 32x32   : 4096 B
constexpr size_t OFF_W2B = 561152;   // end 565248 B

__device__ __forceinline__ float hswish(float v) {
    float t = fminf(fmaxf(v + 3.0f, 0.0f), 6.0f);
    return v * t * (1.0f / 6.0f);
}

// ---- async global->LDS copy (ASYNCcnt path, CDNA5 ISA) ----
__device__ __forceinline__ void async_copy16(const char* gsrc, char* ldst) {
    unsigned int loff = (unsigned int)(uintptr_t)ldst;  // low 32 bits = wave LDS offset
    asm volatile("global_load_async_to_lds_b128 %0, %1, off"
                 :: "v"(loff), "v"(gsrc) : "memory");
}
__device__ __forceinline__ void async_wait0() {
    asm volatile("s_wait_asynccnt 0" ::: "memory");
}

// ---- fragment loaders (wave32 layouts from CDNA5 ISA §7.12.2) ----
// A 16x32 bf16: lane l(0..15) row M=l holds K=[k0..k0+7] comps0-7, K=[k0+16..k0+23] comps8-15;
//               lane 16+l same row, K offset +8 / +24.
__device__ __forceinline__ v16bf load_a_frag(const __bf16* p, int ld, int mb, int k0, int lane) {
    int r  = mb + (lane & 15);
    int kb = k0 + ((lane >> 4) << 3);
    const __bf16* q = p + r * ld + kb;
    v16bf a;
#pragma unroll
    for (int i = 0; i < 8; ++i) { a[i] = q[i]; a[i + 8] = q[i + 16]; }
    return a;
}
// B 32x16 bf16 stored as plane[n][k] (= W[n][k]): lane n comps0-7 = K=[k0..k0+7],
// comps8-15 = K=[k0+16..k0+23]; lane 16+n same, K offset +8 / +24 (matches A).
__device__ __forceinline__ v16bf load_b_frag(const __bf16* p, int ld, int nb, int k0, int lane) {
    int r  = nb + (lane & 15);
    int kb = k0 + ((lane >> 4) << 3);
    const __bf16* q = p + r * ld + kb;
    v16bf b;
#pragma unroll
    for (int i = 0; i < 8; ++i) { b[i] = q[i]; b[i + 8] = q[i + 16]; }
    return b;
}

__device__ __forceinline__ v8f wmma1(v16bf a, v16bf b, v8f c) {
    return __builtin_amdgcn_wmma_f32_16x16x32_bf16(false, a, false, b, (short)0, c, false, false);
}

// Generic LDS GEMM: D(M x N) = A(M x K) * B(K x N), split-bf16 (hi*hi + lo*hi + hi*lo).
// 8 waves; each wave owns one m-strip and NT n-tiles simultaneously: A fragments are
// loaded once per k-chunk (not once per tile), and WMMAs are interleaved pairwise on
// independent accumulators so back-to-back WMMAs have no D->D RAW hazard.
template <int TM, int TN, int KK, int LDA, int LDB, typename StoreF>
__device__ __forceinline__ void wmma_gemm(const __bf16* Ah, const __bf16* Al,
                                          const __bf16* Bh, const __bf16* Bl,
                                          int wave, int lane, StoreF store) {
    constexpr int MS = TM / 16;          // m-strips
    constexpr int NW = 8 / MS;           // waves sharing one strip
    constexpr int NT = (TN / 16) / NW;   // n-tiles per wave
    static_assert(MS * NW == 8 && NT >= 2 && (NT % 2) == 0, "tile mapping");
    const int mb  = (wave % MS) * 16;
    const int nb0 = (wave / MS) * (NT * 16);
    v8f acc[NT];
    const v8f vz = {0, 0, 0, 0, 0, 0, 0, 0};
#pragma unroll
    for (int t = 0; t < NT; ++t) acc[t] = vz;
#pragma unroll
    for (int k0 = 0; k0 < KK; k0 += 32) {
        v16bf aH = load_a_frag(Ah, LDA, mb, k0, lane);
        v16bf aL = load_a_frag(Al, LDA, mb, k0, lane);
#pragma unroll
        for (int t = 0; t < NT; t += 2) {
            v16bf bH0 = load_b_frag(Bh, LDB, nb0 + t * 16, k0, lane);
            v16bf bL0 = load_b_frag(Bl, LDB, nb0 + t * 16, k0, lane);
            v16bf bH1 = load_b_frag(Bh, LDB, nb0 + t * 16 + 16, k0, lane);
            v16bf bL1 = load_b_frag(Bl, LDB, nb0 + t * 16 + 16, k0, lane);
            acc[t]     = wmma1(aH, bH0, acc[t]);
            acc[t + 1] = wmma1(aH, bH1, acc[t + 1]);
            acc[t]     = wmma1(aL, bH0, acc[t]);
            acc[t + 1] = wmma1(aL, bH1, acc[t + 1]);
            acc[t]     = wmma1(aH, bL0, acc[t]);
            acc[t + 1] = wmma1(aH, bL1, acc[t + 1]);
        }
    }
#pragma unroll
    for (int t = 0; t < NT; ++t) store(mb, nb0 + t * 16, acc[t]);
}

// ---------------- weight prep ----------------
__global__ void prep_triu_kernel(const float* __restrict__ W, __bf16* __restrict__ hi, int n) {
    int total = n * n;
    int i = blockIdx.x * 256 + threadIdx.x;
    if (i >= total) return;
    int r = i / n, c = i % n;
    float v = (c <= r) ? W[i] : 0.0f;
    __bf16 h = (__bf16)v;
    hi[i] = h;
    hi[total + i] = (__bf16)(v - (float)h);
}
// Conv weight cw[o][c][j] -> plane[o][j*128+c] (so A rows of x are contiguous K runs).
__global__ void prep_conv_kernel(const float* __restrict__ cw, __bf16* __restrict__ hi, int k) {
    int K = 128 * k;
    int total = 128 * K;
    int i = blockIdx.x * 256 + threadIdx.x;
    if (i >= total) return;
    int o = i / K, kk = i % K;
    int j = kk >> 7, c = kk & 127;
    float v = cw[(o * 128 + c) * k + j];
    __bf16 h = (__bf16)v;
    hi[i] = h;
    hi[total + i] = (__bf16)(v - (float)h);
}

// ---------------- scale 0: LayerNorm + TriU(W0a) + hswish + TriU(W0b), transpose out ----
__global__ __launch_bounds__(256) void scale0_kernel(const float* __restrict__ x,
                                                     const float* __restrict__ gamma,
                                                     const float* __restrict__ beta,
                                                     const float* __restrict__ ba,
                                                     const float* __restrict__ bb,
                                                     const char* __restrict__ ws,
                                                     float* __restrict__ out) {
    extern __shared__ char smem[];
    __bf16* hAh = (__bf16*)smem;        // 128x128
    __bf16* hAl = hAh + 16384;
    __bf16* Wh  = hAl + 16384;          // staged weight planes
    __bf16* Wl  = Wh + 16384;
    __bf16* h1h = Wl + 16384;
    __bf16* h1l = h1h + 16384;
    float*  red = (float*)(h1l + 16384);  // 512 floats
    const int tid = threadIdx.x, lane = tid & 31, wave = tid >> 5;
    const int b = blockIdx.x;
    const float* xb = x + (size_t)b * 16384;

    // async-stage W0a while we do the LayerNorm reduction
    for (int i = tid; i < 4096; i += 256)
        async_copy16((const char*)ws + OFF_W0A + (size_t)i * 16, (char*)Wh + (size_t)i * 16);

    float s = 0.f, ss = 0.f;
    for (int i = tid; i < 16384; i += 256) { float v = xb[i]; s += v; ss += v * v; }
    red[tid] = s; red[256 + tid] = ss;
    __syncthreads();
    for (int off = 128; off > 0; off >>= 1) {
        if (tid < off) { red[tid] += red[tid + off]; red[256 + tid] += red[256 + tid + off]; }
        __syncthreads();
    }
    float mu   = red[0] * (1.0f / 16384.0f);
    float var  = red[256] * (1.0f / 16384.0f) - mu * mu;
    float rstd = rsqrtf(var + 1e-5f);

    for (int i = tid; i < 16384; i += 256) {
        float v = (xb[i] - mu) * rstd * gamma[i] + beta[i];
        __bf16 h = (__bf16)v;
        hAh[i] = h; hAl[i] = (__bf16)(v - (float)h);
    }
    async_wait0();
    __syncthreads();

    wmma_gemm<128, 128, 128, 128, 128>(hAh, hAl, Wh, Wl, wave, lane,
        [&](int mb, int nb, v8f acc) {
            int half = (lane >> 4) & 1, n = nb + (lane & 15);
            float bj = ba[n];
#pragma unroll
            for (int i = 0; i < 8; ++i) {
                int m = mb + half * 8 + i;
                float v = hswish(acc[i] + bj);
                __bf16 h = (__bf16)v;
                h1h[m * 128 + n] = h; h1l[m * 128 + n] = (__bf16)(v - (float)h);
            }
        });
    __syncthreads();
    for (int i = tid; i < 4096; i += 256)
        async_copy16((const char*)ws + OFF_W0B + (size_t)i * 16, (char*)Wh + (size_t)i * 16);
    async_wait0();
    __syncthreads();

    float* outb = out + (size_t)b * 28672;  // rows 0..127: out[b, j, t] = h2[t][j]
    wmma_gemm<128, 128, 128, 128, 128>(h1h, h1l, Wh, Wl, wave, lane,
        [&](int mb, int nb, v8f acc) {
            int half = (lane >> 4) & 1, n = nb + (lane & 15);
            float bj = bb[n];
            int m0 = mb + half * 8;
            float4 lo4 = make_float4(acc[0] + bj, acc[1] + bj, acc[2] + bj, acc[3] + bj);
            float4 hi4 = make_float4(acc[4] + bj, acc[5] + bj, acc[6] + bj, acc[7] + bj);
            *(float4*)(outb + n * 128 + m0)     = lo4;
            *(float4*)(outb + n * 128 + m0 + 4) = hi4;
        });
}

// ---------------- scale 1: conv k=2 (GEMM 64x128x256) + TriU pair over td=64 ----------------
__global__ __launch_bounds__(256) void scale1_kernel(const float* __restrict__ x,
                                                     const float* __restrict__ cb,
                                                     const float* __restrict__ ba,
                                                     const float* __restrict__ bb,
                                                     const char* __restrict__ ws,
                                                     float* __restrict__ out) {
    extern __shared__ char smem[];
    __bf16* Ah  = (__bf16*)smem;    // 64x256 (A rows of x are contiguous)
    __bf16* Al  = Ah + 16384;
    __bf16* Bh  = Al + 16384;       // 128x256 conv weights, later TriU weights
    __bf16* Bl  = Bh + 32768;
    __bf16* hch = Bl + 32768;       // 128x64 (conv out, channel-major)
    __bf16* hcl = hch + 8192;
    __bf16* h1h = hcl + 8192;       // 128x64
    __bf16* h1l = h1h + 8192;
    const int tid = threadIdx.x, lane = tid & 31, wave = tid >> 5;
    const int b = blockIdx.x;
    const float* xb = x + (size_t)b * 16384;

    // async-stage conv weights while converting x to split-bf16
    for (int i = tid; i < 8192; i += 256)
        async_copy16((const char*)ws + OFF_C1 + (size_t)i * 16, (char*)Bh + (size_t)i * 16);
    for (int i = tid; i < 16384; i += 256) {
        float v = xb[i]; __bf16 h = (__bf16)v;
        Ah[i] = h; Al[i] = (__bf16)(v - (float)h);
    }
    async_wait0();
    __syncthreads();

    wmma_gemm<64, 128, 256, 256, 256>(Ah, Al, Bh, Bl, wave, lane,
        [&](int mb, int nb, v8f acc) {
            int half = (lane >> 4) & 1, o = nb + (lane & 15);
            float bo = cb[o];
#pragma unroll
            for (int i = 0; i < 8; ++i) {
                int m = mb + half * 8 + i;
                float v = acc[i] + bo;
                __bf16 h = (__bf16)v;
                hch[o * 64 + m] = h; hcl[o * 64 + m] = (__bf16)(v - (float)h);
            }
        });
    __syncthreads();
    for (int i = tid; i < 1024; i += 256)
        async_copy16((const char*)ws + OFF_W1A + (size_t)i * 16, (char*)Bh + (size_t)i * 16);
    async_wait0();
    __syncthreads();

    wmma_gemm<128, 64, 64, 64, 64>(hch, hcl, Bh, Bh + 4096, wave, lane,
        [&](int mb, int nb, v8f acc) {
            int half = (lane >> 4) & 1, j = nb + (lane & 15);
            float bj = ba[j];
#pragma unroll
            for (int i = 0; i < 8; ++i) {
                int m = mb + half * 8 + i;
                float v = hswish(acc[i] + bj);
                __bf16 h = (__bf16)v;
                h1h[m * 64 + j] = h; h1l[m * 64 + j] = (__bf16)(v - (float)h);
            }
        });
    __syncthreads();
    for (int i = tid; i < 1024; i += 256)
        async_copy16((const char*)ws + OFF_W1B + (size_t)i * 16, (char*)Bh + (size_t)i * 16);
    async_wait0();
    __syncthreads();

    float* outb = out + (size_t)b * 28672 + 16384;  // rows 128..191: out[b,128+j,c]=hc2[c][j]
    wmma_gemm<128, 64, 64, 64, 64>(h1h, h1l, Bh, Bh + 4096, wave, lane,
        [&](int mb, int nb, v8f acc) {
            int half = (lane >> 4) & 1, j = nb + (lane & 15);
            float bj = bb[j];
            int m0 = mb + half * 8;
            float4 lo4 = make_float4(acc[0] + bj, acc[1] + bj, acc[2] + bj, acc[3] + bj);
            float4 hi4 = make_float4(acc[4] + bj, acc[5] + bj, acc[6] + bj, acc[7] + bj);
            *(float4*)(outb + j * 128 + m0)     = lo4;
            *(float4*)(outb + j * 128 + m0 + 4) = hi4;
        });
}

// ---------------- scale 2: conv k=4 (GEMM 32x128x512, K staged in halves) + TriU pair td=32 ----
__global__ __launch_bounds__(256) void scale2_kernel(const float* __restrict__ x,
                                                     const float* __restrict__ cb,
                                                     const float* __restrict__ ba,
                                                     const float* __restrict__ bb,
                                                     const char* __restrict__ ws,
                                                     float* __restrict__ out) {
    extern __shared__ char smem[];
    __bf16* Ah  = (__bf16*)smem;    // 32x512
    __bf16* Al  = Ah + 16384;
    __bf16* Bh  = Al + 16384;       // 128x256 (one K-half of conv weights)
    __bf16* Bl  = Bh + 32768;
    __bf16* hch = Bl + 32768;       // 128x32
    __bf16* hcl = hch + 4096;
    __bf16* h1h = hcl + 4096;       // 128x32
    __bf16* h1l = h1h + 4096;
    const int tid = threadIdx.x, lane = tid & 31, wave = tid >> 5;
    const int b = blockIdx.x;
    const float* xb = x + (size_t)b * 16384;

    for (int i = tid; i < 16384; i += 256) {
        float v = xb[i]; __bf16 h = (__bf16)v;
        Ah[i] = h; Al[i] = (__bf16)(v - (float)h);
    }

    // conv: 16 tiles (2 x 8); wave owns (m=0,n=wave) and (m=1,n=wave); K=512 in two staged halves
    const v8f vz = {0, 0, 0, 0, 0, 0, 0, 0};
    v8f acc0 = vz, acc1 = vz;
    const int nb = wave * 16;
    for (int half = 0; half < 2; ++half) {
        __syncthreads();
        {   // async-stage one K-half of conv2 weights: thread -> (plane p, row r), 512B each
            int p = tid >> 7, r = tid & 127;
            const char* src = (const char*)ws + OFF_C2 + (size_t)p * 131072
                            + ((size_t)r * 64 + (size_t)half * 32) * 16;
            char* dst = (char*)(p ? Bl : Bh) + (size_t)r * 512;
#pragma unroll
            for (int i = 0; i < 32; ++i) async_copy16(src + i * 16, dst + i * 16);
        }
        async_wait0();
        __syncthreads();
#pragma unroll
        for (int k0 = 0; k0 < 256; k0 += 32) {
            v16bf aH0 = load_a_frag(Ah, 512, 0,  half * 256 + k0, lane);
            v16bf aL0 = load_a_frag(Al, 512, 0,  half * 256 + k0, lane);
            v16bf aH1 = load_a_frag(Ah, 512, 16, half * 256 + k0, lane);
            v16bf aL1 = load_a_frag(Al, 512, 16, half * 256 + k0, lane);
            v16bf bH  = load_b_frag(Bh, 256, nb, k0, lane);
            v16bf bL  = load_b_frag(Bl, 256, nb, k0, lane);
            acc0 = wmma1(aH0, bH, acc0);
            acc1 = wmma1(aH1, bH, acc1);
            acc0 = wmma1(aL0, bH, acc0);
            acc1 = wmma1(aL1, bH, acc1);
            acc0 = wmma1(aH0, bL, acc0);
            acc1 = wmma1(aH1, bL, acc1);
        }
    }
    {   // store conv result transposed (channel-major) + bias
        int half = (lane >> 4) & 1, o = nb + (lane & 15);
        float bo = cb[o];
#pragma unroll
        for (int i = 0; i < 8; ++i) {
            int m0 = half * 8 + i;
            float v = acc0[i] + bo;
            __bf16 h = (__bf16)v;
            hch[o * 32 + m0] = h; hcl[o * 32 + m0] = (__bf16)(v - (float)h);
            int m1 = 16 + half * 8 + i;
            v = acc1[i] + bo;
            h = (__bf16)v;
            hch[o * 32 + m1] = h; hcl[o * 32 + m1] = (__bf16)(v - (float)h);
        }
    }
    __syncthreads();
    for (int i = tid; i < 256; i += 256)
        async_copy16((const char*)ws + OFF_W2A + (size_t)i * 16, (char*)Bh + (size_t)i * 16);
    async_wait0();
    __syncthreads();

    wmma_gemm<128, 32, 32, 32, 32>(hch, hcl, Bh, Bh + 1024, wave, lane,
        [&](int mb, int nb2, v8f acc) {
            int half = (lane >> 4) & 1, j = nb2 + (lane & 15);
            float bj = ba[j];
#pragma unroll
            for (int i = 0; i < 8; ++i) {
                int m = mb + half * 8 + i;
                float v = hswish(acc[i] + bj);
                __bf16 h = (__bf16)v;
                h1h[m * 32 + j] = h; h1l[m * 32 + j] = (__bf16)(v - (float)h);
            }
        });
    __syncthreads();
    for (int i = tid; i < 256; i += 256)
        async_copy16((const char*)ws + OFF_W2B + (size_t)i * 16, (char*)Bh + (size_t)i * 16);
    async_wait0();
    __syncthreads();

    float* outb = out + (size_t)b * 28672 + 24576;  // rows 192..223
    wmma_gemm<128, 32, 32, 32, 32>(h1h, h1l, Bh, Bh + 1024, wave, lane,
        [&](int mb, int nb2, v8f acc) {
            int half = (lane >> 4) & 1, j = nb2 + (lane & 15);
            float bj = bb[j];
            int m0 = mb + half * 8;
            float4 lo4 = make_float4(acc[0] + bj, acc[1] + bj, acc[2] + bj, acc[3] + bj);
            float4 hi4 = make_float4(acc[4] + bj, acc[5] + bj, acc[6] + bj, acc[7] + bj);
            *(float4*)(outb + j * 128 + m0)     = lo4;
            *(float4*)(outb + j * 128 + m0 + 4) = hi4;
        });
}

extern "C" void kernel_launch(void* const* d_in, const int* in_sizes, int n_in,
                              void* d_out, int out_size, void* d_ws, size_t ws_size,
                              hipStream_t stream) {
    const float* x     = (const float*)d_in[0];
    const float* gamma = (const float*)d_in[1];
    const float* beta  = (const float*)d_in[2];
    const float* W0a   = (const float*)d_in[3];
    const float* b0a   = (const float*)d_in[4];
    const float* W0b   = (const float*)d_in[5];
    const float* b0b   = (const float*)d_in[6];
    const float* c1w   = (const float*)d_in[7];
    const float* c1b   = (const float*)d_in[8];
    const float* W1a   = (const float*)d_in[9];
    const float* b1a   = (const float*)d_in[10];
    const float* W1b   = (const float*)d_in[11];
    const float* b1b   = (const float*)d_in[12];
    const float* c2w   = (const float*)d_in[13];
    const float* c2b   = (const float*)d_in[14];
    const float* W2a   = (const float*)d_in[15];
    const float* b2a   = (const float*)d_in[16];
    const float* W2b   = (const float*)d_in[17];
    const float* b2b   = (const float*)d_in[18];
    char* ws   = (char*)d_ws;
    float* out = (float*)d_out;

    // weight prep: masked / transposed split-bf16 planes into ws (L2-resident, ~552 KB)
    prep_triu_kernel<<<64,  256, 0, stream>>>(W0a, (__bf16*)(ws + OFF_W0A), 128);
    prep_triu_kernel<<<64,  256, 0, stream>>>(W0b, (__bf16*)(ws + OFF_W0B), 128);
    prep_conv_kernel<<<128, 256, 0, stream>>>(c1w, (__bf16*)(ws + OFF_C1), 2);
    prep_conv_kernel<<<256, 256, 0, stream>>>(c2w, (__bf16*)(ws + OFF_C2), 4);
    prep_triu_kernel<<<16,  256, 0, stream>>>(W1a, (__bf16*)(ws + OFF_W1A), 64);
    prep_triu_kernel<<<16,  256, 0, stream>>>(W1b, (__bf16*)(ws + OFF_W1B), 64);
    prep_triu_kernel<<<4,   256, 0, stream>>>(W2a, (__bf16*)(ws + OFF_W2A), 32);
    prep_triu_kernel<<<4,   256, 0, stream>>>(W2b, (__bf16*)(ws + OFF_W2B), 32);

    // one workgroup (8 wave32) per batch slice; big dynamic LDS (320KB/WGP on CDNA5)
    scale0_kernel<<<2048, 256, 198656, stream>>>(x, gamma, beta, b0a, b0b, ws, out);
    scale1_kernel<<<2048, 256, 262144, stream>>>(x, c1b, b1a, b1b, ws, out);
    scale2_kernel<<<2048, 256, 229376, stream>>>(x, c2b, b2a, b2b, ws, out);
}
